// CrossModalAttention_47562467836047
// MI455X (gfx1250) — matrix operand order repaired and checked by reading, not compiled, not run on hardware
//
#include <hip/hip_runtime.h>
#include <hip/hip_bf16.h>
#include <math.h>

// ---------------------------------------------------------------------------
// CrossModalAttention for MI455X (gfx1250, wave32, WMMA).
// All GEMMs run through v_wmma_f32_16x16x32_bf16 (bf16 in, f32 accumulate).
// Workspace holds bf16 Q/Kt/V pre-swizzled into WMMA fragment-linear layouts
// so the attention inner loop is pure contiguous 32B/lane loads + WMMA.
// Attention processes 32 audio rows per workgroup so every K and V fragment
// is reused across two A-fragments (halves L2 traffic vs 16-row tiles);
// 96KB static LDS leans on CDNA5's 320KB-per-WGP budget.
// ---------------------------------------------------------------------------

typedef __attribute__((ext_vector_type(16))) __bf16 v16bf;
typedef __attribute__((ext_vector_type(8)))  __bf16 v8bf;
typedef __attribute__((ext_vector_type(8)))  float  v8f;
typedef __attribute__((ext_vector_type(4)))  float  v4f;

static __device__ __forceinline__ v8bf ld8bf(const __bf16* p) {
  return *reinterpret_cast<const v8bf*>(p);
}
static __device__ __forceinline__ void st8bf(__bf16* p, v8bf v) {
  *reinterpret_cast<v8bf*>(p) = v;
}
static __device__ __forceinline__ v4f ld4f(const float* p) {
  return *reinterpret_cast<const v4f*>(p);
}
static __device__ __forceinline__ v16bf join16(v8bf a, v8bf b) {
  return __builtin_shufflevector(a, b, 0,1,2,3,4,5,6,7,8,9,10,11,12,13,14,15);
}
// B-fragment: 16 contiguous bf16 per lane (pre-swizzled layout)
static __device__ __forceinline__ v16bf ldB(const __bf16* p) {
  return join16(ld8bf(p), ld8bf(p + 8));
}
static __device__ __forceinline__ v8f wmma_bf16(v16bf a, v16bf b, v8f c) {
  // (neg_a, A, neg_b, B, c_mod, C, reuse_a, reuse_b)
  return __builtin_amdgcn_wmma_f32_16x16x32_bf16(false, a, false, b, (short)0, c,
                                                 false, false);
}

// ---------------------------------------------------------------------------
// Kernel 1: swizzle a f32 weight matrix W[K][512] into bf16 WMMA-B fragment
// layout: block (nTile 0..31, kc 0..K/32-1), per lane 16 contiguous values:
//   Wsw[(nTile*(K/32)+kc)*512 + lane*16 + j]
//     = W[kc*32 + (lane<16?0:8) + (j<8?j:j+8)][nTile*16 + lane%16]
// ---------------------------------------------------------------------------
__global__ __launch_bounds__(256) void wswz_kernel(const float* __restrict__ W,
                                                   int Kdim,
                                                   __hip_bfloat16* __restrict__ dstp) {
  __bf16* dst = reinterpret_cast<__bf16*>(dstp);
  const int lane = threadIdx.x & 31;
  const int wid  = threadIdx.x >> 5;
  const int kchunks = Kdim >> 5;
  const int tile = blockIdx.x * 8 + wid;          // tile = nTile*kchunks + kc
  if (tile >= 32 * kchunks) return;
  const int nTile = tile / kchunks;
  const int kc    = tile % kchunks;
  const int o16   = (lane < 16) ? 0 : 8;
  const int col   = nTile * 16 + (lane & 15);
  v8bf lo, hi;
#pragma unroll
  for (int j = 0; j < 8; ++j) {
    lo[j] = (__bf16)W[(size_t)(kc * 32 + o16 + j) * 512 + col];
    hi[j] = (__bf16)W[(size_t)(kc * 32 + o16 + 16 + j) * 512 + col];
  }
  __bf16* d = dst + (size_t)tile * 512 + lane * 16;
  st8bf(d, lo);
  st8bf(d + 8, hi);
}

// ---------------------------------------------------------------------------
// Kernel 2: projection Y = X*W + b in bf16 via WMMA, output pre-swizzled.
//   mode 0 (Q): 16x32 tile -> A-fragment layout
//   mode 1 (K): 16x32 tile -> B-fragment layout of K^T
//   mode 2 (V): 32x16 tile -> B-fragment layout of V
// One wave per tile, 8 waves per block. Tile staged in per-wave LDS (same-wave
// DS ops are in-order, no barrier needed), then re-read in swizzled order.
// ---------------------------------------------------------------------------
__global__ __launch_bounds__(256) void proj_kernel(const float* __restrict__ X, int Kdim,
                                                   const __hip_bfloat16* __restrict__ Wswp,
                                                   const float* __restrict__ bias,
                                                   __hip_bfloat16* __restrict__ dstp,
                                                   int mode, int nTiles) {
  __shared__ __bf16 tl[8][16 * 32];
  const __bf16* Wsw = reinterpret_cast<const __bf16*>(Wswp);
  __bf16* dst = reinterpret_cast<__bf16*>(dstp);

  const int lane = threadIdx.x & 31;
  const int wid  = threadIdx.x >> 5;
  const int tileId = blockIdx.x * 8 + wid;
  if (tileId >= nTiles) return;
  const int kchunks = Kdim >> 5;
  const int m15 = lane & 15;
  const int oA  = (lane < 16) ? 0 : 8;    // A-fragment half-wave offset
  const int oB  = (lane < 16) ? 0 : 16;   // unused; B fragments here use the same 0/8 offset
  const v8f z = {0.f, 0.f, 0.f, 0.f, 0.f, 0.f, 0.f, 0.f};
  __bf16* t = tl[wid];

  if (mode < 2) {
    const int rowTile = tileId >> 4;      // 16 output rows
    const int cChunk  = tileId & 15;      // 32 output cols (h)
    const float* xrow = X + (size_t)(rowTile * 16 + m15) * Kdim;
    const __bf16* w0 = Wsw + (size_t)((cChunk * 2 + 0) * kchunks) * 512 + lane * 16;
    const __bf16* w1 = Wsw + (size_t)((cChunk * 2 + 1) * kchunks) * 512 + lane * 16;
    v8f c0 = z, c1 = z;
    for (int kc = 0; kc < kchunks; ++kc) {
      const float* p = xrow + kc * 32 + oA;
      v4f f0 = ld4f(p), f1 = ld4f(p + 4), f2 = ld4f(p + 16), f3 = ld4f(p + 20);
      v16bf a;
#pragma unroll
      for (int i = 0; i < 4; ++i) {
        a[i]      = (__bf16)f0[i];
        a[4 + i]  = (__bf16)f1[i];
        a[8 + i]  = (__bf16)f2[i];
        a[12 + i] = (__bf16)f3[i];
      }
      v16bf b0 = ldB(w0 + (size_t)kc * 512);
      v16bf b1 = ldB(w1 + (size_t)kc * 512);
      c0 = wmma_bf16(a, b0, c0);
      c1 = wmma_bf16(a, b1, c1);
    }
    const float bb0 = bias[cChunk * 32 + m15];
    const float bb1 = bias[cChunk * 32 + 16 + m15];
#pragma unroll
    for (int r = 0; r < 8; ++r) {
      const int m = r + oA;                   // D layout: row = r (+8 upper half)
      t[m * 32 + m15]      = (__bf16)(c0[r] + bb0);
      t[m * 32 + 16 + m15] = (__bf16)(c1[r] + bb1);
    }
    const int off0 = m15 * 32 + oA;
    const int off1 = m15 * 32 + 16 + oA;
    v8bf p0 = *(const v8bf*)&t[off0];
    v8bf p1 = *(const v8bf*)&t[off1];
    __bf16* d = dst + (size_t)tileId * 512 + lane * 16;
    st8bf(d, p0);
    st8bf(d + 8, p1);
  } else {
    // V: 32 rows (t) x 16 cols (h); stage transposed [h][t] for B-swizzle read.
    const int rowTile32 = tileId >> 5;
    const int nTile     = tileId & 31;
    const float* xlo = X + (size_t)(rowTile32 * 32 + m15) * Kdim;
    const float* xhi = xlo + (size_t)16 * Kdim;
    const __bf16* w = Wsw + (size_t)(nTile * kchunks) * 512 + lane * 16;
    v8f clo = z, chi = z;
    for (int kc = 0; kc < kchunks; ++kc) {
      v16bf alo, ahi;
      {
        const float* p = xlo + kc * 32 + oA;
        v4f f0 = ld4f(p), f1 = ld4f(p + 4), f2 = ld4f(p + 16), f3 = ld4f(p + 20);
#pragma unroll
        for (int i = 0; i < 4; ++i) {
          alo[i] = (__bf16)f0[i]; alo[4 + i] = (__bf16)f1[i];
          alo[8 + i] = (__bf16)f2[i]; alo[12 + i] = (__bf16)f3[i];
        }
      }
      {
        const float* p = xhi + kc * 32 + oA;
        v4f f0 = ld4f(p), f1 = ld4f(p + 4), f2 = ld4f(p + 16), f3 = ld4f(p + 20);
#pragma unroll
        for (int i = 0; i < 4; ++i) {
          ahi[i] = (__bf16)f0[i]; ahi[4 + i] = (__bf16)f1[i];
          ahi[8 + i] = (__bf16)f2[i]; ahi[12 + i] = (__bf16)f3[i];
        }
      }
      v16bf bb = ldB(w + (size_t)kc * 512);
      clo = wmma_bf16(alo, bb, clo);
      chi = wmma_bf16(ahi, bb, chi);
    }
    const float bv = bias[nTile * 16 + m15];
#pragma unroll
    for (int r = 0; r < 8; ++r) {
      const int mlo = r + oA;
      const int mhi = 16 + r + oA;
      t[m15 * 32 + mlo] = (__bf16)(clo[r] + bv);
      t[m15 * 32 + mhi] = (__bf16)(chi[r] + bv);
    }
    v8bf p0 = *(const v8bf*)&t[m15 * 32 + oA];
    v8bf p1 = *(const v8bf*)&t[m15 * 32 + oA + 16];
    const int b  = rowTile32 >> 4;          // 16 t-chunks of 32 per batch
    const int tc = rowTile32 & 15;
    __bf16* d = dst + ((size_t)((b * 32 + nTile) * 16 + tc)) * 512 + lane * 16;
    st8bf(d, p0);
    st8bf(d + 8, p1);
  }
}

// ---------------------------------------------------------------------------
// Kernel 3: fused attention. One workgroup (8 waves) per 32 audio rows
// (two 16-row M-tiles). Wave w owns score/output columns [w*64, w*64+64).
// Each K fragment feeds 2 score A-frags; each V fragment feeds 2 attn A-frags.
// LDS: 32KB q block + 64KB f32 scores (32x512); softmax scratch overlays the
// q buffer (dead after the score GEMMs).
// ---------------------------------------------------------------------------
__global__ __launch_bounds__(256) void attn_kernel(const __hip_bfloat16* __restrict__ Qswp,
                                                   const __hip_bfloat16* __restrict__ Ktswp,
                                                   const __hip_bfloat16* __restrict__ Vswp,
                                                   const int* __restrict__ mask,
                                                   float* __restrict__ out) {
  __shared__ __bf16 qs[32 * 512];       // q block, A-fragment-linear (32 KB)
  __shared__ float  sc[32 * 512];       // scores / exp(scores) (64 KB)
  float* red = reinterpret_cast<float*>(qs);   // overlay: 32x8 partials
  float* rs  = red + 256;                      // overlay: 32 row sums

  const __bf16* Qsw  = reinterpret_cast<const __bf16*>(Qswp);
  const __bf16* Ktsw = reinterpret_cast<const __bf16*>(Ktswp);
  const __bf16* Vsw  = reinterpret_cast<const __bf16*>(Vswp);

  const int tid  = threadIdx.x;
  const int lane = tid & 31;
  const int w    = tid >> 5;
  const int mT32 = blockIdx.x;                // 0..511 (32 rows each)
  const int b    = mT32 >> 6;                 // 64 row-blocks per batch
  const int row0 = (mT32 & 63) << 5;
  const int m15  = lane & 15;
  const int oA   = (lane < 16) ? 0 : 8;
  const v8f z = {0.f, 0.f, 0.f, 0.f, 0.f, 0.f, 0.f, 0.f};

  // --- stage 32-row q block (fragment-linear -> flat 32KB copy) ---
  {
    const v8bf* src = (const v8bf*)(Qsw + (size_t)mT32 * 16384);
    v8bf* dq = (v8bf*)qs;
#pragma unroll
    for (int i = 0; i < 8; ++i) dq[tid + i * 256] = src[tid + i * 256];
  }
  __syncthreads();

  // --- scores: 2 M-tiles x 4 N-tiles per wave; K fragments loaded once ---
  v8f sA0 = z, sA1 = z, sA2 = z, sA3 = z;     // rows row0..row0+15
  v8f sB0 = z, sB1 = z, sB2 = z, sB3 = z;     // rows row0+16..row0+31
  const __bf16* kb = Ktsw + ((size_t)(b * 32 + w * 4) * 16) * 512 + lane * 16;
  for (int kc = 0; kc < 16; ++kc) {
    const __bf16* kp = kb + (size_t)kc * 512;
    __builtin_prefetch((const void*)(kp + 512), 0, 3);   // next k-chunk -> WGP$
    const __bf16* qp = qs + kc * 512 + lane * 16;
    v16bf aA = join16(ld8bf(qp), ld8bf(qp + 8));
    v16bf aB = join16(ld8bf(qp + 8192), ld8bf(qp + 8192 + 8));
    v16bf b0 = ldB(kp);
    v16bf b1 = ldB(kp + 8192);
    v16bf b2 = ldB(kp + 16384);
    v16bf b3 = ldB(kp + 24576);
    sA0 = wmma_bf16(aA, b0, sA0);  sB0 = wmma_bf16(aB, b0, sB0);
    sA1 = wmma_bf16(aA, b1, sA1);  sB1 = wmma_bf16(aB, b1, sB1);
    sA2 = wmma_bf16(aA, b2, sA2);  sB2 = wmma_bf16(aB, b2, sB2);
    sA3 = wmma_bf16(aA, b3, sA3);  sB3 = wmma_bf16(aB, b3, sB3);
  }
  // scale + mask -> LDS (mask is per score-column == per lane: cheap)
  const float scale = 0.0441941738241592f;     // 512^-0.5
  const int colb = w * 64 + m15;
  const int mv0 = mask[b * 512 + colb];
  const int mv1 = mask[b * 512 + colb + 16];
  const int mv2 = mask[b * 512 + colb + 32];
  const int mv3 = mask[b * 512 + colb + 48];
#pragma unroll
  for (int r = 0; r < 8; ++r) {
    const int m = r + oA;
    float* sr0 = &sc[m * 512 + colb];
    sr0[0]  = mv0 ? sA0[r] * scale : -1e30f;
    sr0[16] = mv1 ? sA1[r] * scale : -1e30f;
    sr0[32] = mv2 ? sA2[r] * scale : -1e30f;
    sr0[48] = mv3 ? sA3[r] * scale : -1e30f;
    float* sr1 = &sc[(16 + m) * 512 + colb];
    sr1[0]  = mv0 ? sB0[r] * scale : -1e30f;
    sr1[16] = mv1 ? sB1[r] * scale : -1e30f;
    sr1[32] = mv2 ? sB2[r] * scale : -1e30f;
    sr1[48] = mv3 ? sB3[r] * scale : -1e30f;
  }
  __syncthreads();   // also retires all q reads before the 'red' overlay is used

  // --- softmax over 512 columns: 8 threads per row, 32 rows ---
  const int srow = tid >> 3, sub = tid & 7;
  float mx = -3.0e38f;
  for (int j = 0; j < 64; ++j) mx = fmaxf(mx, sc[srow * 512 + sub + j * 8]);
  red[srow * 8 + sub] = mx;
  __syncthreads();
  float rmax = red[srow * 8];
#pragma unroll
  for (int i = 1; i < 8; ++i) rmax = fmaxf(rmax, red[srow * 8 + i]);
  __syncthreads();
  float ps = 0.f;
  for (int j = 0; j < 64; ++j) {
    const int idx = srow * 512 + sub + j * 8;
    const float e = __expf(sc[idx] - rmax);
    sc[idx] = e;
    ps += e;
  }
  red[srow * 8 + sub] = ps;
  __syncthreads();
  if (sub == 0) {
    float s = 0.f;
#pragma unroll
    for (int i = 0; i < 8; ++i) s += red[srow * 8 + i];
    rs[srow] = s;
  }
  __syncthreads();

  // --- out = softmax(scores) . V ; V fragments loaded once for both M-tiles;
  //     1/rowsum folded into the A-frag conversion (row == lane%16) ---
  const float rinvA = 1.0f / rs[m15];
  const float rinvB = 1.0f / rs[16 + m15];
  v8f oA0 = z, oA1 = z, oA2 = z, oA3 = z;
  v8f oB0 = z, oB1 = z, oB2 = z, oB3 = z;
  const __bf16* vb = Vsw + ((size_t)(b * 32 + w * 4) * 16) * 512 + lane * 16;
  for (int tc = 0; tc < 16; ++tc) {
    const __bf16* vp = vb + (size_t)tc * 512;
    __builtin_prefetch((const void*)(vp + 512), 0, 3);   // next t-chunk -> WGP$
    v16bf aA, aB;
    {
      const float* ap = &sc[m15 * 512 + tc * 32 + oA];
      v4f f0 = ld4f(ap), f1 = ld4f(ap + 4), f2 = ld4f(ap + 16), f3 = ld4f(ap + 20);
#pragma unroll
      for (int i = 0; i < 4; ++i) {
        aA[i]      = (__bf16)(f0[i] * rinvA);
        aA[4 + i]  = (__bf16)(f1[i] * rinvA);
        aA[8 + i]  = (__bf16)(f2[i] * rinvA);
        aA[12 + i] = (__bf16)(f3[i] * rinvA);
      }
    }
    {
      const float* ap = &sc[(16 + m15) * 512 + tc * 32 + oA];
      v4f f0 = ld4f(ap), f1 = ld4f(ap + 4), f2 = ld4f(ap + 16), f3 = ld4f(ap + 20);
#pragma unroll
      for (int i = 0; i < 4; ++i) {
        aB[i]      = (__bf16)(f0[i] * rinvB);
        aB[4 + i]  = (__bf16)(f1[i] * rinvB);
        aB[8 + i]  = (__bf16)(f2[i] * rinvB);
        aB[12 + i] = (__bf16)(f3[i] * rinvB);
      }
    }
    v16bf b0 = ldB(vp);
    v16bf b1 = ldB(vp + 8192);
    v16bf b2 = ldB(vp + 16384);
    v16bf b3 = ldB(vp + 24576);
    oA0 = wmma_bf16(aA, b0, oA0);  oB0 = wmma_bf16(aB, b0, oB0);
    oA1 = wmma_bf16(aA, b1, oA1);  oB1 = wmma_bf16(aB, b1, oB1);
    oA2 = wmma_bf16(aA, b2, oA2);  oB2 = wmma_bf16(aB, b2, oB2);
    oA3 = wmma_bf16(aA, b3, oA3);  oB3 = wmma_bf16(aB, b3, oB3);
  }
  // store D fragments: lane col h = w*64 + tile*16 + m15, rows r(+8)
  float* op = out + ((size_t)(b * 2048 + row0)) * 512 + colb;
#pragma unroll
  for (int r = 0; r < 8; ++r) {
    const int m = r + oA;
    float* orow0 = op + (size_t)m * 512;
    orow0[0]  = oA0[r];
    orow0[16] = oA1[r];
    orow0[32] = oA2[r];
    orow0[48] = oA3[r];
    float* orow1 = op + (size_t)(16 + m) * 512;
    orow1[0]  = oB0[r];
    orow1[16] = oB1[r];
    orow1[32] = oB2[r];
    orow1[48] = oB3[r];
  }
}

// ---------------------------------------------------------------------------
extern "C" void kernel_launch(void* const* d_in, const int* in_sizes, int n_in,
                              void* d_out, int out_size, void* d_ws, size_t ws_size,
                              hipStream_t stream) {
  (void)in_sizes; (void)n_in; (void)out_size; (void)ws_size;
  const float* audio = (const float*)d_in[0];   // (8,2048,512)
  const float* text  = (const float*)d_in[1];   // (8,512,768)
  const float* Wq    = (const float*)d_in[2];
  const float* bq    = (const float*)d_in[3];
  const float* Wk    = (const float*)d_in[4];
  const float* bk    = (const float*)d_in[5];
  const float* Wv    = (const float*)d_in[6];
  const float* bv    = (const float*)d_in[7];
  const int*   mask  = (const int*)d_in[8];     // (8,512)
  float* out = (float*)d_out;

  char* ws = (char*)d_ws;
  size_t off = 0;
  __hip_bfloat16* Qsw  = (__hip_bfloat16*)(ws + off); off += (size_t)16384 * 512 * 2;
  __hip_bfloat16* Ktsw = (__hip_bfloat16*)(ws + off); off += (size_t)8 * 512 * 512 * 2;
  __hip_bfloat16* Vsw  = (__hip_bfloat16*)(ws + off); off += (size_t)8 * 512 * 512 * 2;
  __hip_bfloat16* WqS  = (__hip_bfloat16*)(ws + off); off += (size_t)512 * 512 * 2;
  __hip_bfloat16* WkS  = (__hip_bfloat16*)(ws + off); off += (size_t)768 * 512 * 2;
  __hip_bfloat16* WvS  = (__hip_bfloat16*)(ws + off); off += (size_t)768 * 512 * 2;

  // 1) weight swizzles (tiny)
  wswz_kernel<<<64, 256, 0, stream>>>(Wq, 512, WqS);   // 32*16 waves
  wswz_kernel<<<96, 256, 0, stream>>>(Wk, 768, WkS);   // 32*24 waves
  wswz_kernel<<<96, 256, 0, stream>>>(Wv, 768, WvS);

  // 2) projections (one wave per tile, 8 waves/block)
  proj_kernel<<<2048, 256, 0, stream>>>(audio, 512, WqS, bq, Qsw, 0, 16384);
  proj_kernel<<<512, 256, 0, stream>>>(text, 768, WkS, bk, Ktsw, 1, 4096);
  proj_kernel<<<512, 256, 0, stream>>>(text, 768, WvS, bv, Vsw, 2, 4096);

  // 3) fused attention: one WG per 32 audio rows
  attn_kernel<<<512, 256, 0, stream>>>(Qsw, Ktsw, Vsw, mask, out);
}
